// VectorQuantizer_77206332113562
// MI455X (gfx1250) — matrix operand order repaired and checked by reading, not compiled, not run on hardware
//
#include <hip/hip_runtime.h>
#include <hip/hip_bf16.h>

typedef __attribute__((ext_vector_type(16))) __bf16 v16bf;
typedef __attribute__((ext_vector_type(8)))  float  v8f;

#define B_   32
#define C_   64
#define H_   64
#define W_   64
#define K_   512
#define HW_  (H_ * W_)          // 4096
#define N_   (B_ * H_ * W_)     // 131072
#define NZQ  (B_ * C_ * HW_)    // 8388608
#define TM   128                // pixel rows per workgroup
#define NWG  (N_ / TM)          // 1024
#define IDX_OFF  NZQ
#define LOSS_OFF (NZQ + N_)
#define ZPITCH 72               // padded LDS row pitch (ushorts), 144B = 9*16B

__device__ __forceinline__ unsigned short f32_to_bf16_rne(float f) {
    unsigned u = __float_as_uint(f);
    u += 0x7FFFu + ((u >> 16) & 1u);
    return (unsigned short)(u >> 16);
}
__device__ __forceinline__ float bf16_to_f32(unsigned short s) {
    return __uint_as_float(((unsigned)s) << 16);
}

// d_ws layout (bytes):
//   [0,        65536)  codebook bf16 [512][64]
//   [65536,    67584)  enorm f32 [512]
//   [67584,    71680)  per-workgroup loss partials f32 [1024]

__global__ void vq_prep(const float* __restrict__ cb,
                        unsigned short* __restrict__ cbh,
                        float* __restrict__ enorm) {
    int code = blockIdx.x * blockDim.x + threadIdx.x;
    if (code >= K_) return;
    float s = 0.f;
    #pragma unroll 4
    for (int c = 0; c < C_; ++c) {
        unsigned short h = f32_to_bf16_rne(cb[code * C_ + c]);
        cbh[code * C_ + c] = h;
        float f = bf16_to_f32(h);
        s += f * f;
    }
    enorm[code] = s;
}

union Frag { v16bf v; uint4 q[2]; };

__global__ __launch_bounds__(256)
void vq_main(const float* __restrict__ ze, const float* __restrict__ cb,
             const unsigned short* __restrict__ cbh,
             const float* __restrict__ enormG,
             float* __restrict__ out, float* __restrict__ partial) {
    __shared__ __align__(16) unsigned short zt[TM * ZPITCH]; // bf16 z tile, [row][chan]
    __shared__ float enorm[K_];
    __shared__ int   bidx[TM];
    __shared__ float red[256];

    const int t   = threadIdx.x;
    const int n0  = blockIdx.x * TM;
    const int b   = n0 >> 12;          // 4096 pixels per batch image
    const int pix = n0 & (HW_ - 1);    // 128-aligned => 2 full W-lines
    const float* zbase = ze + (size_t)b * C_ * HW_ + pix;

    for (int i = t; i < K_; i += 256) enorm[i] = enormG[i];

    // Stage + transpose z tile ([C,HW] strided -> [row][chan] bf16 in LDS)
    #pragma unroll
    for (int k = 0; k < (TM * C_) / 256; ++k) {
        int i  = t + k * 256;
        int c  = i >> 7;          // channel
        int hw = i & (TM - 1);    // pixel within tile
        zt[hw * ZPITCH + c] = f32_to_bf16_rne(zbase[(size_t)c * HW_ + hw]);
    }
    __syncthreads();

    const int lane = t & 31, wv = t >> 5;
    const int half = lane >> 4, rl = lane & 15;
    const int rowT = wv * 16;          // this wave's 16-row tile

    // A fragments (16x32 bf16, K chunks c*32): per ISA layout, lane holds
    // chans {c*32 + half*8 + 0..7} and {c*32 + 16 + half*8 + 0..7} of row rl.
    Frag a0, a1;
    const unsigned short* rp = &zt[(rowT + rl) * ZPITCH];
    a0.q[0] = *(const uint4*)(rp + half * 8);
    a0.q[1] = *(const uint4*)(rp + 16 + half * 8);
    a1.q[0] = *(const uint4*)(rp + 32 + half * 8);
    a1.q[1] = *(const uint4*)(rp + 48 + half * 8);

    float bd[8]; int bi[8];
    #pragma unroll
    for (int r = 0; r < 8; ++r) { bd[r] = 3.4e38f; bi[r] = 0x7fffffff; }

    for (int j = 0; j < K_ / 16; ++j) {
        // B fragment (32x16 bf16): lane = code column rl of tile;
        // chans {c*32 + half*8 + 0..7} and {c*32 + 16 + half*8 + 0..7} of codebook row (j*16 + rl).
        Frag b0, b1;
        const unsigned short* cp = cbh + (size_t)(j * 16 + rl) * C_ + half * 8;
        b0.q[0] = *(const uint4*)(cp);
        b0.q[1] = *(const uint4*)(cp + 16);
        b1.q[0] = *(const uint4*)(cp + 32);
        b1.q[1] = *(const uint4*)(cp + 48);

        v8f acc = {};
        acc = __builtin_amdgcn_wmma_f32_16x16x32_bf16(false, a0.v, false, b0.v,
                                                      (short)0, acc, false, false);
        acc = __builtin_amdgcn_wmma_f32_16x16x32_bf16(false, a1.v, false, b1.v,
                                                      (short)0, acc, false, false);
        const int code = j * 16 + rl;
        const float en = enorm[code];
        #pragma unroll
        for (int r = 0; r < 8; ++r) {
            float d = fmaf(-2.f, acc[r], en);   // ||e||^2 - 2 z.e
            if (d < bd[r]) { bd[r] = d; bi[r] = code; }
        }
    }

    // Reduce over the 16 lanes (same half) holding the same row; lowest-index ties.
    #pragma unroll
    for (int off = 1; off < 16; off <<= 1) {
        #pragma unroll
        for (int r = 0; r < 8; ++r) {
            float od = __shfl_xor(bd[r], off, 32);
            int   oi = __shfl_xor(bi[r], off, 32);
            if (od < bd[r] || (od == bd[r] && oi < bi[r])) { bd[r] = od; bi[r] = oi; }
        }
    }
    if (rl == 0) {
        #pragma unroll
        for (int r = 0; r < 8; ++r) bidx[rowT + half * 8 + r] = bi[r];
    }
    __syncthreads();

    // encoding_indices (as float, per output convention)
    if (t < TM) out[IDX_OFF + n0 + t] = (float)bidx[t];

    // z_q_st (== z_q forward) from exact f32 codebook + loss partial
    float* zq = out + (size_t)b * C_ * HW_ + pix;
    float lsum = 0.f;
    #pragma unroll
    for (int k = 0; k < (TM * C_) / 256; ++k) {
        int i  = t + k * 256;
        int c  = i >> 7;
        int hw = i & (TM - 1);
        float q = cb[bidx[hw] * C_ + c];
        size_t o = (size_t)c * HW_ + hw;
        float d = q - zbase[o];
        zq[o] = q;
        lsum += d * d;
    }
    red[t] = lsum;
    __syncthreads();
    for (int off = 128; off > 0; off >>= 1) {
        if (t < off) red[t] += red[t + off];
        __syncthreads();
    }
    if (t == 0) partial[blockIdx.x] = red[0];
}

__global__ void vq_loss(const float* __restrict__ partial, float* __restrict__ out) {
    __shared__ float red[256];
    int t = threadIdx.x;
    float s = 0.f;
    for (int i = t; i < NWG; i += 256) s += partial[i];
    red[t] = s;
    __syncthreads();
    for (int off = 128; off > 0; off >>= 1) {
        if (t < off) red[t] += red[t + off];
        __syncthreads();
    }
    // loss = q_loss + 0.25*e_loss, and q_loss == e_loss in forward value
    if (t == 0) out[LOSS_OFF] = 1.25f * red[0] / (float)NZQ;
}

extern "C" void kernel_launch(void* const* d_in, const int* in_sizes, int n_in,
                              void* d_out, int out_size, void* d_ws, size_t ws_size,
                              hipStream_t stream) {
    const float* ze = (const float*)d_in[0];   // [32,64,64,64] f32
    const float* cb = (const float*)d_in[1];   // [512,64] f32
    float* out = (float*)d_out;

    unsigned short* cbh   = (unsigned short*)d_ws;
    float* enorm          = (float*)((char*)d_ws + (size_t)K_ * C_ * 2);
    float* partial        = (float*)((char*)d_ws + (size_t)K_ * C_ * 2 + K_ * 4);

    vq_prep<<<2, 256, 0, stream>>>(cb, cbh, enorm);
    vq_main<<<NWG, 256, 0, stream>>>(ze, cb, cbh, enorm, out, partial);
    vq_loss<<<1, 256, 0, stream>>>(partial, out);
}
